// HAN_12575664243207
// MI455X (gfx1250) — matrix operand order repaired and checked by reading, not compiled, not run on hardware
//
#include <hip/hip_runtime.h>
#include <hip/hip_bf16.h>
#include <math.h>

// ---------------- problem constants (from reference setup_inputs) -----------
#define BB    4       // batch
#define MM    5       // metapaths
#define NN    1000    // nodes
#define FIN   64
#define HH    4       // heads
#define DD    64      // per-head dim
#define HD    256     // HH*DD
#define SHID  128
#define NOUT  5
#define THRESH 0.97f

typedef __attribute__((ext_vector_type(16))) __bf16        v16bf;
typedef __attribute__((ext_vector_type(8)))  float         v8f;
typedef __attribute__((ext_vector_type(4)))  unsigned int  v4u;
typedef __attribute__((ext_vector_type(8)))  unsigned int  v8u;

// ---------------------------------------------------------------------------
// Convert x[:,0] (B,M,N,Fin) f32 -> hbf (B,N,Fin) bf16
__global__ __launch_bounds__(256) void conv_x_bf16(const float* __restrict__ x,
                                                   __bf16* __restrict__ out) {
  int idx = blockIdx.x * 256 + threadIdx.x;
  const int total = BB * NN * FIN;
  if (idx >= total) return;
  int k = idx % FIN;
  int n = (idx / FIN) % NN;
  int b = idx / (NN * FIN);
  out[idx] = (__bf16)x[(((size_t)b * MM + 0) * NN + n) * FIN + k];
}

// Convert f32 buffer -> bf16 (for layer-2 activations)
__global__ __launch_bounds__(256) void conv_f32_bf16(const float* __restrict__ in,
                                                     __bf16* __restrict__ out, int total) {
  int idx = blockIdx.x * 256 + threadIdx.x;
  if (idx < total) out[idx] = (__bf16)in[idx];
}

// Transpose + convert weights: W (M, K, P) f32 -> Wt (M, P, K) bf16
__global__ __launch_bounds__(256) void prep_w_bf16(const float* __restrict__ W,
                                                   __bf16* __restrict__ Wt, int K, int P) {
  int idx = blockIdx.x * 256 + threadIdx.x;
  int total = MM * K * P;
  if (idx >= total) return;
  int k = idx % K;
  int p = (idx / K) % P;
  int m = idx / (K * P);
  Wt[idx] = (__bf16)W[((size_t)m * K + k) * P + p];
}

// ---------------------------------------------------------------------------
// WMMA GEMM: C[z] = A[b] @ B[m]^T-stored, f32 accumulate.
//   A  : (BB, rows, K) bf16 row-major        (z = b*MM + m)
//   Bt : (MM, P, K)   bf16 (B transposed)
//   C  : (BB*MM, rows, P) f32
// One 16x16 tile per wave; 8 waves per block cover 128 output columns.
// Edge rows are handled by CLAMPING the A row (branch-free K-loop, EXEC stays
// all-ones around v_wmma) and guarding only the final stores.
// grid = (ceil(rows/16), P/128, BB*MM), block = 256.
__global__ __launch_bounds__(256) void gemm_bf16_wmma(const __bf16* __restrict__ A,
                                                      const __bf16* __restrict__ Bt,
                                                      float* __restrict__ C,
                                                      int rows, int K, int P) {
  int z = blockIdx.z;
  int b = z / MM, m = z % MM;
  const __bf16* Ab  = A  + (size_t)b * rows * K;
  const __bf16* Btm = Bt + (size_t)m * P * K;
  float*        Cz  = C  + (size_t)z * rows * P;

  int wave = threadIdx.x >> 5;
  int lane = threadIdx.x & 31;
  int half = lane >> 4;       // K-half selector
  int l16  = lane & 15;

  int row0 = blockIdx.x * 16;
  int col0 = (blockIdx.y * 8 + wave) * 16;

  int arow = row0 + l16;                // A row for this lane
  if (arow >= rows) arow = rows - 1;    // clamp: OOB rows compute, never store
  int akb  = half * 8;                  // A: elems 0-7 -> K=akb..akb+7, 8-15 -> +16
  int bcol = col0 + l16;                // B column for this lane
  int bkb  = half * 8;                  // B: elems 0-7 -> K=bkb..bkb+7, 8-15 -> +16

  const __bf16* ap = Ab  + (size_t)arow * K + akb;
  const __bf16* bp = Btm + (size_t)bcol * K + bkb;

  v8f acc = {};
  for (int k0 = 0; k0 < K; k0 += 32) {
    v4u lo = *reinterpret_cast<const v4u*>(ap + k0);        // K = k0+akb .. +7
    v4u hi = *reinterpret_cast<const v4u*>(ap + k0 + 16);   // K = k0+akb+16 .. +23
    v8u au;
#pragma unroll
    for (int i = 0; i < 4; ++i) { au[i] = lo[i]; au[4 + i] = hi[i]; }
    v16bf af = __builtin_bit_cast(v16bf, au);

    v4u blo = *reinterpret_cast<const v4u*>(bp + k0);
    v4u bhi = *reinterpret_cast<const v4u*>(bp + k0 + 16);
    v8u bu;
#pragma unroll
    for (int i = 0; i < 4; ++i) { bu[i] = blo[i]; bu[4 + i] = bhi[i]; }
    v16bf bf = __builtin_bit_cast(v16bf, bu);

    acc = __builtin_amdgcn_wmma_f32_16x16x32_bf16(false, af, false, bf,
                                                  (short)0, acc, false, false);
  }

#pragma unroll
  for (int v = 0; v < 8; ++v) {
    int r = row0 + half * 8 + v;        // C layout: lane>=16 holds M=8..15
    if (r < rows) Cz[(size_t)r * P + col0 + l16] = acc[v];
  }
}

// ---------------------------------------------------------------------------
// el/er head projections: el[z,n,h] = sum_d feat[z,n,h*64+d] * al[m,h,d]
__global__ __launch_bounds__(256) void el_er_kernel(const float* __restrict__ feat,
                                                    const float* __restrict__ al,
                                                    const float* __restrict__ ar,
                                                    float* __restrict__ el,
                                                    float* __restrict__ er) {
  int idx = blockIdx.x * 256 + threadIdx.x;
  const int total = BB * MM * NN * HH;
  if (idx >= total) return;
  int h = idx & 3;
  int n = (idx >> 2) % NN;
  int z = idx / (NN * HH);
  int m = z % MM;
  const float* f  = feat + ((size_t)z * NN + n) * HD + h * DD;
  const float* a1 = al + (m * HH + h) * DD;
  const float* a2 = ar + (m * HH + h) * DD;
  float s1 = 0.f, s2 = 0.f;
#pragma unroll 8
  for (int d = 0; d < DD; ++d) { s1 += f[d] * a1[d]; s2 += f[d] * a2[d]; }
  el[idx] = s1;
  er[idx] = s2;
}

// ---------------------------------------------------------------------------
// GAT attention + aggregation. One block per (dst, z=b*MM+m).
// Pass1: online masked softmax stats over sources (block reduce).
// Pass2: per-source weights to LDS, then thread-per-feature accumulation.
__global__ __launch_bounds__(256) void gat_attention(const float* __restrict__ adj,
                                                     const int*   __restrict__ node_nums,
                                                     const float* __restrict__ feat,
                                                     const float* __restrict__ el,
                                                     const float* __restrict__ er,
                                                     const float* __restrict__ bias,
                                                     float* __restrict__ zout) {
  __shared__ float red[256 * 8];
  __shared__ float wbuf[NN * HH];

  int dst = blockIdx.x;
  int z   = blockIdx.y;
  int b   = z / MM, m = z % MM;
  int tid = threadIdx.x;
  int nn  = node_nums[b];
  bool dstOK = (dst < nn);

  const float* adjcol = adj + (size_t)z * NN * NN + dst;   // stride NN over s
  const float* elz    = el + (size_t)z * NN * HH;
  float erh[HH];
  {
    const float* e = er + ((size_t)z * NN + dst) * HH;
#pragma unroll
    for (int h = 0; h < HH; ++h) erh[h] = e[h];
  }

  // ---- pass 1: online softmax stats -----------------------------------
  float mx[HH] = {-3.0e38f, -3.0e38f, -3.0e38f, -3.0e38f};
  float sm[HH] = {0.f, 0.f, 0.f, 0.f};
  for (int s = tid; s < NN; s += 256) {
    if (s + 256 < NN) __builtin_prefetch(&adjcol[(size_t)(s + 256) * NN], 0, 1);
    float av = adjcol[(size_t)s * NN];
    if (dstOK && av > THRESH && s < nn) {
#pragma unroll
      for (int h = 0; h < HH; ++h) {
        float e = elz[s * HH + h] + erh[h];
        e = (e > 0.f) ? e : 0.2f * e;            // leaky_relu(0.2)
        float mo = mx[h];
        float mn = fmaxf(mo, e);
        sm[h] = sm[h] * __expf(mo - mn) + __expf(e - mn);
        mx[h] = mn;
      }
    }
  }
#pragma unroll
  for (int h = 0; h < HH; ++h) { red[tid * 8 + h] = mx[h]; red[tid * 8 + 4 + h] = sm[h]; }
  __syncthreads();
  for (int off = 128; off > 0; off >>= 1) {
    if (tid < off) {
#pragma unroll
      for (int h = 0; h < HH; ++h) {
        float m1 = red[tid * 8 + h],        l1 = red[tid * 8 + 4 + h];
        float m2 = red[(tid + off) * 8 + h], l2 = red[(tid + off) * 8 + 4 + h];
        float mmx = fmaxf(m1, m2);
        red[tid * 8 + h]     = mmx;
        red[tid * 8 + 4 + h] = l1 * __expf(m1 - mmx) + l2 * __expf(m2 - mmx);
      }
    }
    __syncthreads();
  }
  float MX[HH], SM[HH];
#pragma unroll
  for (int h = 0; h < HH; ++h) { MX[h] = red[h]; SM[h] = red[4 + h]; }
  __syncthreads();

  // ---- pass 2a: per-source weights into LDS ---------------------------
  for (int s = tid; s < NN; s += 256) {
    float av = adjcol[(size_t)s * NN];                    // L2 hit (re-read)
    bool ms = dstOK && (av > THRESH) && (s < nn);
#pragma unroll
    for (int h = 0; h < HH; ++h) {
      float w = 0.f;
      if (ms && SM[h] > 0.f) {
        float e = elz[s * HH + h] + erh[h];
        e = (e > 0.f) ? e : 0.2f * e;
        w = __expf(e - MX[h]) / SM[h];
      }
      wbuf[s * HH + h] = w;
    }
  }
  __syncthreads();

  // ---- pass 2b: weighted aggregation, thread per output feature -------
  int f = tid;                 // 0..255
  int h = f >> 6;              // head uniform per wave (32 threads)
  float acc = 0.f;
  const float* fz = feat + (size_t)z * NN * HD + f;
  for (int s = 0; s < NN; ++s) {
    float w = wbuf[s * HH + h];
    if (w != 0.f) acc += w * fz[(size_t)s * HD];
  }
  float r   = acc + bias[m * HD + f];
  float emb = (r > 0.f) ? r : (__expf(r) - 1.f);          // elu
  zout[(((size_t)b * NN + dst) * MM + m) * HD + f] = emb;
}

// ---------------------------------------------------------------------------
// Semantic attention part 1: p[b,n,m] = tanh(z_row @ W1 + b1) @ W2, masked.
// One block (128 threads) per (n, m, b). Deterministic partials to pbuf.
__global__ __launch_bounds__(128) void sem_part1(const float* __restrict__ z,
                                                 const int*   __restrict__ node_nums,
                                                 const float* __restrict__ W1,
                                                 const float* __restrict__ b1,
                                                 const float* __restrict__ W2,
                                                 float* __restrict__ pbuf) {
  __shared__ float zrow[HD];
  __shared__ float red[SHID];
  int n = blockIdx.x, m = blockIdx.y, b = blockIdx.z;
  int j = threadIdx.x;
  const float* zp = z + (((size_t)b * NN + n) * MM + m) * HD;
  zrow[j]         = zp[j];
  zrow[j + SHID]  = zp[j + SHID];
  __syncthreads();
  float dot = b1[j];
#pragma unroll 8
  for (int d = 0; d < HD; ++d) dot += zrow[d] * W1[d * SHID + j];  // coalesced in j
  red[j] = tanhf(dot) * W2[j];
  __syncthreads();
  for (int off = 64; off > 0; off >>= 1) {
    if (j < off) red[j] += red[j + off];
    __syncthreads();
  }
  if (j == 0) {
    int nn = node_nums[b];
    pbuf[((size_t)b * NN + n) * MM + m] = (n < nn) ? red[0] : 0.f;
  }
}

// Semantic attention part 2: beta = softmax_m( sum_{b,n} pbuf / sum_b nn[b] )
__global__ __launch_bounds__(256) void sem_beta(const float* __restrict__ pbuf,
                                                const int*   __restrict__ node_nums,
                                                float* __restrict__ beta) {
  __shared__ float red[256];
  __shared__ float wsum[MM];
  int tid = threadIdx.x;
  for (int m = 0; m < MM; ++m) {
    float s = 0.f;
    for (int i = tid; i < BB * NN; i += 256) s += pbuf[(size_t)i * MM + m];
    red[tid] = s;
    __syncthreads();
    for (int off = 128; off > 0; off >>= 1) {
      if (tid < off) red[tid] += red[tid + off];
      __syncthreads();
    }
    if (tid == 0) wsum[m] = red[0];
    __syncthreads();
  }
  if (tid == 0) {
    float nsum = 0.f;
    for (int b = 0; b < BB; ++b) nsum += (float)node_nums[b];
    float w[MM], mxv = -3.0e38f;
    for (int m = 0; m < MM; ++m) { w[m] = wsum[m] / nsum; mxv = fmaxf(mxv, w[m]); }
    float s = 0.f;
    for (int m = 0; m < MM; ++m) { w[m] = __expf(w[m] - mxv); s += w[m]; }
    for (int m = 0; m < MM; ++m) beta[m] = w[m] / s;
  }
}

// h2[b,n,f] = sum_m beta[m] * z[b,n,m,f]
__global__ __launch_bounds__(256) void combine_kernel(const float* __restrict__ z,
                                                      const float* __restrict__ beta,
                                                      float* __restrict__ h2) {
  int idx = blockIdx.x * 256 + threadIdx.x;
  const int total = BB * NN * HD;
  if (idx >= total) return;
  int f = idx & (HD - 1);
  int n = (idx / HD) % NN;
  int b = idx / (NN * HD);
  float s = 0.f;
#pragma unroll
  for (int m = 0; m < MM; ++m)
    s += beta[m] * z[(((size_t)b * NN + n) * MM + m) * HD + f];
  h2[idx] = s;
}

// logits[b,n,o] = (h2_row @ pred_W + pred_b)[o] * nmask
__global__ __launch_bounds__(256) void pred_kernel(const float* __restrict__ h2,
                                                   const float* __restrict__ W,
                                                   const float* __restrict__ bias,
                                                   const int*   __restrict__ node_nums,
                                                   float* __restrict__ out) {
  int idx = blockIdx.x * 256 + threadIdx.x;
  const int total = BB * NN * NOUT;
  if (idx >= total) return;
  int o = idx % NOUT;
  int n = (idx / NOUT) % NN;
  int b = idx / (NN * NOUT);
  int nn = node_nums[b];
  float s = bias[o];
  const float* hp = h2 + ((size_t)b * NN + n) * HD;
#pragma unroll 8
  for (int d = 0; d < HD; ++d) s += hp[d] * W[d * NOUT + o];
  out[idx] = (n < nn) ? s : 0.f;
}

// ---------------------------------------------------------------------------
extern "C" void kernel_launch(void* const* d_in, const int* in_sizes, int n_in,
                              void* d_out, int out_size, void* d_ws, size_t ws_size,
                              hipStream_t stream) {
  const float* x        = (const float*)d_in[0];
  const float* adj      = (const float*)d_in[1];
  const int*   nnums    = (const int*)  d_in[2];
  const float* fc1      = (const float*)d_in[3];
  const float* al1      = (const float*)d_in[4];
  const float* ar1      = (const float*)d_in[5];
  const float* gb1      = (const float*)d_in[6];
  const float* sW1_1    = (const float*)d_in[7];
  const float* sb1_1    = (const float*)d_in[8];
  const float* sW2_1    = (const float*)d_in[9];
  const float* fc2      = (const float*)d_in[10];
  const float* al2      = (const float*)d_in[11];
  const float* ar2      = (const float*)d_in[12];
  const float* gb2      = (const float*)d_in[13];
  const float* sW1_2    = (const float*)d_in[14];
  const float* sb1_2    = (const float*)d_in[15];
  const float* sW2_2    = (const float*)d_in[16];
  const float* predW    = (const float*)d_in[17];
  const float* predb    = (const float*)d_in[18];
  float* out = (float*)d_out;

  // ---- workspace layout (bytes, 256-aligned) ----
  char* ws = (char*)d_ws;
  __bf16* hbf  = (__bf16*)(ws + 0);                         //  2,048,000 B
  __bf16* wt   = (__bf16*)(ws + 2048000);                   //    655,360 B
  float*  feat = (float*) (ws + 2703360);                   // 20,480,000 B
  float*  el   = (float*) (ws + 23183360);                  //    320,000 B
  float*  er   = (float*) (ws + 23503360);                  //    320,000 B
  float*  zbuf = (float*) (ws + 23823360);                  // 20,480,000 B
  float*  h2   = (float*) (ws + 44303360);                  //  4,096,000 B
  float*  pbuf = (float*) (ws + 48399360);                  //    320,000 B
  float*  beta = (float*) (ws + 48719360);                  //        256 B
  (void)ws_size; // requires ~48.8 MB of scratch

  dim3 blk256(256), blk128(128);

  // ================= Layer 1 =================
  conv_x_bf16<<<(BB * NN * FIN + 255) / 256, blk256, 0, stream>>>(x, hbf);
  prep_w_bf16<<<(MM * FIN * HD + 255) / 256, blk256, 0, stream>>>(fc1, wt, FIN, HD);
  gemm_bf16_wmma<<<dim3((NN + 15) / 16, HD / 128, BB * MM), blk256, 0, stream>>>(
      hbf, wt, feat, NN, FIN, HD);
  el_er_kernel<<<(BB * MM * NN * HH + 255) / 256, blk256, 0, stream>>>(feat, al1, ar1, el, er);
  gat_attention<<<dim3(NN, BB * MM), blk256, 0, stream>>>(adj, nnums, feat, el, er, gb1, zbuf);
  sem_part1<<<dim3(NN, MM, BB), blk128, 0, stream>>>(zbuf, nnums, sW1_1, sb1_1, sW2_1, pbuf);
  sem_beta<<<1, blk256, 0, stream>>>(pbuf, nnums, beta);
  combine_kernel<<<(BB * NN * HD + 255) / 256, blk256, 0, stream>>>(zbuf, beta, h2);

  // ================= Layer 2 =================
  conv_f32_bf16<<<(BB * NN * HD + 255) / 256, blk256, 0, stream>>>(h2, hbf, BB * NN * HD);
  prep_w_bf16<<<(MM * HD * HD + 255) / 256, blk256, 0, stream>>>(fc2, wt, HD, HD);
  gemm_bf16_wmma<<<dim3((NN + 15) / 16, HD / 128, BB * MM), blk256, 0, stream>>>(
      hbf, wt, feat, NN, HD, HD);
  el_er_kernel<<<(BB * MM * NN * HH + 255) / 256, blk256, 0, stream>>>(feat, al2, ar2, el, er);
  gat_attention<<<dim3(NN, BB * MM), blk256, 0, stream>>>(adj, nnums, feat, el, er, gb2, zbuf);
  sem_part1<<<dim3(NN, MM, BB), blk128, 0, stream>>>(zbuf, nnums, sW1_2, sb1_2, sW2_2, pbuf);
  sem_beta<<<1, blk256, 0, stream>>>(pbuf, nnums, beta);
  combine_kernel<<<(BB * NN * HD + 255) / 256, blk256, 0, stream>>>(zbuf, beta, h2);

  // ================= Prediction head =================
  pred_kernel<<<(BB * NN * NOUT + 255) / 256, blk256, 0, stream>>>(h2, predW, predb, nnums, out);
}
